// MixtralBlock_5523327943199
// MI455X (gfx1250) — compile-verified, matrix operand order checked by reading, not run on hardware
//
#include <hip/hip_runtime.h>
#include <hip/hip_bf16.h>

// ---------------------------------------------------------------------------
// MI455X (gfx1250) Mixtral block: bf16 WMMA GEMMs everywhere, f32 accumulate.
// wave32, WMMA 16x16x32 bf16. Block tile 128x128x32, 8 waves, 32x64 per wave,
// software-pipelined global->reg->LDS staging + global_prefetch_b8 on weights.
// ---------------------------------------------------------------------------

enum { S = 2048, Dm = 1024, Hn = 8, KHn = 4, HDn = 128, En = 8, FFn = 3584 };

typedef __attribute__((ext_vector_type(16))) __bf16 v16bf;
typedef __attribute__((ext_vector_type(8)))  float  v8f;
typedef __attribute__((ext_vector_type(4)))  unsigned int v4u;
typedef __attribute__((ext_vector_type(4)))  float  v4f;

union FragU { v4u q[2]; v16bf v; };

__device__ __forceinline__ v4u zero4() { v4u z = {0, 0, 0, 0}; return z; }

__device__ __forceinline__ unsigned short f2bf(float f) {
  union { float f; unsigned u; } v; v.f = f;
  unsigned r = v.u + 0x7FFFu + ((v.u >> 16) & 1u);
  return (unsigned short)(r >> 16);
}
__device__ __forceinline__ float bf2f(unsigned short h) {
  union { unsigned u; float f; } v; v.u = ((unsigned)h) << 16;
  return v.f;
}

// ---------------------------------------------------------------------------
// Generic WMMA GEMM: C[M,N] = op(A[M,K] bf16 x B[K,N]) with options:
//   BF32      : B is f32 (weights, converted to bf16 on the fly) else bf16
//   Mptr      : device-side M (per-expert token count), null -> host M
//   idxA      : gather A rows (token list), null -> identity
//   idxC      : scatter C rows (+gate scale), null -> identity
//   addSrc    : residual added in epilogue (direct C-row indexing)
//   gate      : per-token scale, gate[idxC[row]*gateStride]
//   accumulate: C += val instead of C = val
//   causalSkip: skip blocks fully above the diagonal (QK^T)
// ---------------------------------------------------------------------------
#define LSTR 40  // padded LDS row stride (ushorts) to dodge bank conflicts

template <int BF32>
__global__ __launch_bounds__(256) void gemm_wmma(
    const unsigned short* __restrict__ A, int lda,
    const void* __restrict__ Bp, int ldb,
    float* __restrict__ C, int ldc,
    int M, int N, int K,
    const int* __restrict__ Mptr,
    const int* __restrict__ idxA,
    const int* __restrict__ idxC,
    const float* __restrict__ addSrc,
    const float* __restrict__ gate, int gateStride,
    float alpha, int accumulate, int causalSkip)
{
  if (Mptr) M = *Mptr;
  int mBase = blockIdx.y * 128;
  if (mBase >= M) return;
  int nBase = blockIdx.x * 128;
  if (causalSkip && nBase > mBase + 127) return;

  __shared__ __attribute__((aligned(16))) unsigned short sA[128][LSTR];
  __shared__ __attribute__((aligned(16))) unsigned short sB[128][LSTR]; // [n][k]

  int tid  = threadIdx.x;
  int lane = tid & 31, wave = tid >> 5;
  int waveM = wave >> 1, waveN = wave & 1;     // 4 x 2 wave grid -> 32 x 64
  int half = lane >> 4, l16 = lane & 15;

  v8f acc[2][4] = {};

  // A-tile loader: each thread owns half a 32-elem row of the 128x32 tile
  int r  = tid >> 1;             // 0..127
  int ca = (tid & 1) << 4;       // 0 or 16
  int garow = mBase + r;
  long arowg = -1;
  if (garow < M) arowg = idxA ? idxA[garow] : garow;

  // B-tile loader: 16 contiguous cols of one k-row, stored transposed
  int kr = tid >> 3;             // 0..31  (k within tile)
  int cb = (tid & 7) << 4;       // 0..112 (n within tile)

  // register staging for software pipeline
  v4u a0 = zero4(), a1 = zero4();
  unsigned short tb[16];

  auto loadTile = [&](int kb) {
    a0 = zero4(); a1 = zero4();
    if (arowg >= 0) {
      const v4u* pa = (const v4u*)(A + (size_t)arowg * lda + kb + ca);
      a0 = pa[0];
      a1 = pa[1];
    }
    if (BF32) {
      const float* Bf = (const float*)Bp;
      const v4f* pb = (const v4f*)(Bf + (size_t)(kb + kr) * ldb + nBase + cb);
      v4f b0 = pb[0], b1 = pb[1], b2 = pb[2], b3 = pb[3];
      tb[0]  = f2bf(b0.x); tb[1]  = f2bf(b0.y); tb[2]  = f2bf(b0.z); tb[3]  = f2bf(b0.w);
      tb[4]  = f2bf(b1.x); tb[5]  = f2bf(b1.y); tb[6]  = f2bf(b1.z); tb[7]  = f2bf(b1.w);
      tb[8]  = f2bf(b2.x); tb[9]  = f2bf(b2.y); tb[10] = f2bf(b2.z); tb[11] = f2bf(b2.w);
      tb[12] = f2bf(b3.x); tb[13] = f2bf(b3.y); tb[14] = f2bf(b3.z); tb[15] = f2bf(b3.w);
    } else {
      const unsigned short* Bh = (const unsigned short*)Bp;
      const v4u* pb = (const v4u*)(Bh + (size_t)(kb + kr) * ldb + nBase + cb);
      v4u b0 = pb[0], b1 = pb[1];
      tb[0]  = b0.x & 0xffff; tb[1]  = b0.x >> 16;
      tb[2]  = b0.y & 0xffff; tb[3]  = b0.y >> 16;
      tb[4]  = b0.z & 0xffff; tb[5]  = b0.z >> 16;
      tb[6]  = b0.w & 0xffff; tb[7]  = b0.w >> 16;
      tb[8]  = b1.x & 0xffff; tb[9]  = b1.x >> 16;
      tb[10] = b1.y & 0xffff; tb[11] = b1.y >> 16;
      tb[12] = b1.z & 0xffff; tb[13] = b1.z >> 16;
      tb[14] = b1.w & 0xffff; tb[15] = b1.w >> 16;
    }
  };

  loadTile(0);

  for (int kb = 0; kb < K; kb += 32) {
    __syncthreads();  // previous iteration's LDS reads complete
    *(v4u*)&sA[r][ca]     = a0;
    *(v4u*)&sA[r][ca + 8] = a1;
#pragma unroll
    for (int j = 0; j < 16; ++j) sB[cb + j][kr] = tb[j];
    __syncthreads();

    if (kb + 32 < K) {
      if (kb + 64 < K) {
        // CDNA5 prefetch hint (global_prefetch_b8) on the weight stream
        const char* pf = BF32
            ? (const char*)((const float*)Bp + (size_t)(kb + 64 + kr) * ldb + nBase + cb)
            : (const char*)((const unsigned short*)Bp + (size_t)(kb + 64 + kr) * ldb + nBase + cb);
        __builtin_prefetch(pf, 0, 3);
      }
      loadTile(kb + 32);  // overlap next-tile global loads with WMMA below
    }

    // ISA 7.12.2 bf16 fragment layout:
    // lane<16:  elems 0..7 = K 0..7,  elems 8..15 = K 16..23
    // lane>=16: elems 0..7 = K 8..15, elems 8..15 = K 24..31
    FragU af[2], bfr[4];
#pragma unroll
    for (int i = 0; i < 2; ++i) {
      int ar = waveM * 32 + i * 16 + l16;
      af[i].q[0] = *(const v4u*)&sA[ar][8 * half];
      af[i].q[1] = *(const v4u*)&sA[ar][16 + 8 * half];
    }
#pragma unroll
    for (int j = 0; j < 4; ++j) {
      int bc = waveN * 64 + j * 16 + l16;
      bfr[j].q[0] = *(const v4u*)&sB[bc][8 * half];
      bfr[j].q[1] = *(const v4u*)&sB[bc][16 + 8 * half];
    }
#pragma unroll
    for (int i = 0; i < 2; ++i)
#pragma unroll
      for (int j = 0; j < 4; ++j)
        acc[i][j] = __builtin_amdgcn_wmma_f32_16x16x32_bf16(
            false, af[i].v, false, bfr[j].v, (short)0, acc[i][j], false, false);
  }

  // Epilogue. C/D layout: VGPR v, lanes 0-15 -> (M=v, N=lane); 16-31 -> (M=v+8).
#pragma unroll
  for (int i = 0; i < 2; ++i) {
#pragma unroll
    for (int j = 0; j < 4; ++j) {
      int colg = nBase + waveN * 64 + j * 16 + l16;
#pragma unroll
      for (int v = 0; v < 8; ++v) {
        int rl = mBase + waveM * 32 + i * 16 + v + 8 * half;
        if (rl < M) {
          float val = acc[i][j][v] * alpha;
          long crow = idxC ? idxC[rl] : rl;
          if (gate) val *= gate[(size_t)crow * gateStride];
          size_t co = (size_t)crow * ldc + colg;
          if (addSrc) val += addSrc[co];
          if (accumulate) C[co] += val; else C[co] = val;
        }
      }
    }
  }
}

// ---------------------------------------------------------------------------
// Elementwise / reduction kernels
// ---------------------------------------------------------------------------
__global__ void rmsnorm_bf16(const float* __restrict__ x, const float* __restrict__ w,
                             unsigned short* __restrict__ out, int ncols) {
  int row = blockIdx.x, tid = threadIdx.x;
  __shared__ float red[256];
  const float* xr = x + (size_t)row * ncols;
  float ss = 0.f;
  for (int i = tid; i < ncols; i += 256) { float v = xr[i]; ss += v * v; }
  red[tid] = ss; __syncthreads();
  for (int s = 128; s > 0; s >>= 1) { if (tid < s) red[tid] += red[tid + s]; __syncthreads(); }
  float inv = rsqrtf(red[0] / ncols + 1e-5f);
  for (int i = tid; i < ncols; i += 256)
    out[(size_t)row * ncols + i] = f2bf(xr[i] * inv * w[i]);
}

__global__ void rope_cast(const float* __restrict__ q, const float* __restrict__ k,
                          const float* __restrict__ v,
                          unsigned short* __restrict__ qb,
                          unsigned short* __restrict__ kT,   // [KH][HD][S]
                          unsigned short* __restrict__ vb) {
  int s = blockIdx.x, tid = threadIdx.x;
  for (int p = tid; p < Hn * HDn / 2; p += 256) {
    int head = p >> 6, i = p & 63;
    float freq = __powf(1.0e6f, -((float)(2 * i) / 128.0f));
    float sn, cs; __sincosf((float)s * freq, &sn, &cs);
    size_t base = (size_t)s * 1024 + head * 128 + 2 * i;
    float x1 = q[base], x2 = q[base + 1];
    qb[base]     = f2bf(x1 * cs - x2 * sn);
    qb[base + 1] = f2bf(x1 * sn + x2 * cs);
  }
  for (int p = tid; p < KHn * HDn / 2; p += 256) {
    int kh = p >> 6, i = p & 63;
    float freq = __powf(1.0e6f, -((float)(2 * i) / 128.0f));
    float sn, cs; __sincosf((float)s * freq, &sn, &cs);
    size_t base = (size_t)s * 512 + kh * 128 + 2 * i;
    float x1 = k[base], x2 = k[base + 1];
    size_t tb = (size_t)kh * HDn * S + (size_t)(2 * i) * S + s;
    kT[tb]     = f2bf(x1 * cs - x2 * sn);
    kT[tb + S] = f2bf(x1 * sn + x2 * cs);
  }
  for (int j = tid; j < 512; j += 256) {
    size_t base = (size_t)s * 512 + j;
    vb[base] = f2bf(v[base]);
  }
}

__global__ void softmax_causal(const float* __restrict__ scores,
                               unsigned short* __restrict__ probs, float scale) {
  int r = blockIdx.x, tid = threadIdx.x;
  __shared__ float red[256];
  const float* sr = scores + (size_t)r * S;
  float mx = -3.0e38f;
  for (int j = tid; j <= r; j += 256) mx = fmaxf(mx, sr[j] * scale);
  red[tid] = mx; __syncthreads();
  for (int s = 128; s > 0; s >>= 1) { if (tid < s) red[tid] = fmaxf(red[tid], red[tid + s]); __syncthreads(); }
  mx = red[0]; __syncthreads();
  float sum = 0.f;
  for (int j = tid; j <= r; j += 256) sum += __expf(sr[j] * scale - mx);
  red[tid] = sum; __syncthreads();
  for (int s = 128; s > 0; s >>= 1) { if (tid < s) red[tid] += red[tid + s]; __syncthreads(); }
  float inv = 1.0f / red[0];
  unsigned short* pr = probs + (size_t)r * S;
  for (int j = tid; j < S; j += 256)
    pr[j] = (j <= r) ? f2bf(__expf(sr[j] * scale - mx) * inv) : (unsigned short)0;
}

__global__ void router_topk(const unsigned short* __restrict__ xn,
                            const float* __restrict__ rw,
                            float* __restrict__ dense_w) {
  int t = blockIdx.x, tid = threadIdx.x;
  float acc[En] = {};
  for (int d = tid; d < Dm; d += 256) {
    float xv = bf2f(xn[(size_t)t * Dm + d]);
#pragma unroll
    for (int e = 0; e < En; ++e) acc[e] += xv * rw[(size_t)d * En + e];
  }
  __shared__ float red[256];
  __shared__ float logits[En];
  for (int e = 0; e < En; ++e) {
    red[tid] = acc[e]; __syncthreads();
    for (int s = 128; s > 0; s >>= 1) { if (tid < s) red[tid] += red[tid + s]; __syncthreads(); }
    if (tid == 0) logits[e] = red[0];
    __syncthreads();
  }
  if (tid == 0) {
    int e1 = 0;
    for (int e = 1; e < En; ++e) if (logits[e] > logits[e1]) e1 = e;
    int e2 = -1;
    for (int e = 0; e < En; ++e) { if (e == e1) continue; if (e2 < 0 || logits[e] > logits[e2]) e2 = e; }
    float g1 = 1.0f / (1.0f + __expf(logits[e2] - logits[e1]));
    for (int e = 0; e < En; ++e) dense_w[(size_t)t * En + e] = 0.0f;
    dense_w[(size_t)t * En + e1] = g1;
    dense_w[(size_t)t * En + e2] = 1.0f - g1;
  }
}

// Deterministic per-expert token compaction (prefix scan, one block per expert)
__global__ void expert_assign(const float* __restrict__ dense_w,
                              int* __restrict__ counts, int* __restrict__ idx) {
  int e = blockIdx.x, tid = threadIdx.x;
  __shared__ int scan[256];
  __shared__ int base;
  if (tid == 0) base = 0;
  __syncthreads();
  for (int t0 = 0; t0 < S; t0 += 256) {
    int t = t0 + tid;
    int f = (dense_w[(size_t)t * En + e] > 0.0f) ? 1 : 0;
    scan[tid] = f; __syncthreads();
    for (int off = 1; off < 256; off <<= 1) {
      int v = (tid >= off) ? scan[tid - off] : 0;
      __syncthreads();
      scan[tid] += v;
      __syncthreads();
    }
    if (f) idx[(size_t)e * S + base + scan[tid] - 1] = t;
    __syncthreads();
    if (tid == 0) base += scan[255];
    __syncthreads();
  }
  if (tid == 0) counts[e] = base;
}

__global__ void silu_mul(const float* __restrict__ g, const float* __restrict__ u,
                         unsigned short* __restrict__ h, const int* __restrict__ Mptr) {
  int row = blockIdx.y;
  if (row >= *Mptr) return;
  int i = blockIdx.x * 256 + threadIdx.x;
  size_t o = (size_t)row * FFn + i;
  float gv = g[o];
  h[o] = f2bf(gv / (1.0f + __expf(-gv)) * u[o]);
}

__global__ void cast_f32_bf16(const float* __restrict__ in, unsigned short* __restrict__ out, int n) {
  int i = blockIdx.x * 256 + threadIdx.x;
  if (i < n) out[i] = f2bf(in[i]);
}
__global__ void copy_f32(const float* __restrict__ in, float* __restrict__ out, int n) {
  int i = blockIdx.x * 256 + threadIdx.x;
  if (i < n) out[i] = in[i];
}

// ---------------------------------------------------------------------------
extern "C" void kernel_launch(void* const* d_in, const int* in_sizes, int n_in,
                              void* d_out, int out_size, void* d_ws, size_t ws_size,
                              hipStream_t stream) {
  (void)in_sizes; (void)n_in; (void)out_size; (void)ws_size;
  const float* x        = (const float*)d_in[0];
  const float* ln_w     = (const float*)d_in[1];
  const float* wq       = (const float*)d_in[2];
  const float* wk       = (const float*)d_in[3];
  const float* wv       = (const float*)d_in[4];
  const float* wo       = (const float*)d_in[5];
  const float* ff_ln_w  = (const float*)d_in[6];
  const float* router_w = (const float*)d_in[7];
  const float* w1       = (const float*)d_in[8];
  const float* w3       = (const float*)d_in[9];
  const float* w2       = (const float*)d_in[10];
  float* out = (float*)d_out;

  char* ws = (char*)d_ws;
  size_t off = 0;
  auto alloc = [&](size_t b) -> void* {
    void* p = ws + off;
    off += (b + 255) & ~(size_t)255;
    return p;
  };

  unsigned short* xn_bf  = (unsigned short*)alloc((size_t)S * Dm * 2);
  float*          qf     = (float*)alloc((size_t)S * 1024 * 4);
  float*          kf     = (float*)alloc((size_t)S * 512 * 4);
  float*          vf     = (float*)alloc((size_t)S * 512 * 4);
  unsigned short* q_bf   = (unsigned short*)alloc((size_t)S * 1024 * 2);
  unsigned short* kT_bf  = (unsigned short*)alloc((size_t)KHn * HDn * S * 2);
  unsigned short* v_bf   = (unsigned short*)alloc((size_t)S * 512 * 2);
  float*          scores = (float*)alloc((size_t)S * S * 4);
  unsigned short* p_bf   = (unsigned short*)alloc((size_t)S * S * 2);
  float*          ctxf   = (float*)alloc((size_t)S * 1024 * 4);
  unsigned short* ctx_bf = (unsigned short*)alloc((size_t)S * 1024 * 2);
  float*          x2     = (float*)alloc((size_t)S * Dm * 4);
  unsigned short* xn2_bf = (unsigned short*)alloc((size_t)S * Dm * 2);
  float*          dnsw   = (float*)alloc((size_t)S * En * 4);
  int*            counts = (int*)alloc((size_t)En * 4);
  int*            idxb   = (int*)alloc((size_t)En * S * 4);
  float*          gbuf   = (float*)alloc((size_t)S * FFn * 4);
  float*          ubuf   = (float*)alloc((size_t)S * FFn * 4);
  unsigned short* h_bf   = (unsigned short*)alloc((size_t)S * FFn * 2);

  dim3 T(256);
  const float one = 1.0f;

  // 1) RMSNorm -> bf16
  rmsnorm_bf16<<<S, T, 0, stream>>>(x, ln_w, xn_bf, Dm);

  // 2) Q/K/V projections (A bf16, B f32 weights)
  gemm_wmma<1><<<dim3(1024 / 128, S / 128), T, 0, stream>>>(
      xn_bf, Dm, wq, 1024, qf, 1024, S, 1024, Dm,
      nullptr, nullptr, nullptr, nullptr, nullptr, 0, one, 0, 0);
  gemm_wmma<1><<<dim3(512 / 128, S / 128), T, 0, stream>>>(
      xn_bf, Dm, wk, 512, kf, 512, S, 512, Dm,
      nullptr, nullptr, nullptr, nullptr, nullptr, 0, one, 0, 0);
  gemm_wmma<1><<<dim3(512 / 128, S / 128), T, 0, stream>>>(
      xn_bf, Dm, wv, 512, vf, 512, S, 512, Dm,
      nullptr, nullptr, nullptr, nullptr, nullptr, 0, one, 0, 0);

  // 3) RoPE + cast (K written transposed [KH][HD][S] for plain QK^T GEMM)
  rope_cast<<<S, T, 0, stream>>>(qf, kf, vf, q_bf, kT_bf, v_bf);

  // 4) Attention per head: QK^T (causal-skipped) -> softmax -> P*V
  const float iscale = 0.08838834764831845f; // 1/sqrt(128)
  for (int h = 0; h < Hn; ++h) {
    int kh = h >> 1;
    gemm_wmma<0><<<dim3(S / 128, S / 128), T, 0, stream>>>(
        q_bf + h * HDn, Hn * HDn, kT_bf + (size_t)kh * HDn * S, S,
        scores, S, S, S, HDn,
        nullptr, nullptr, nullptr, nullptr, nullptr, 0, one, 0, /*causal*/1);
    softmax_causal<<<S, T, 0, stream>>>(scores, p_bf, iscale);
    gemm_wmma<0><<<dim3(HDn / 128, S / 128), T, 0, stream>>>(
        p_bf, S, v_bf + kh * HDn, KHn * HDn,
        ctxf + h * HDn, Hn * HDn, S, HDn, S,
        nullptr, nullptr, nullptr, nullptr, nullptr, 0, one, 0, 0);
  }

  // 5) Output projection + residual
  cast_f32_bf16<<<(S * 1024) / 256, T, 0, stream>>>(ctxf, ctx_bf, S * 1024);
  gemm_wmma<1><<<dim3(1024 / 128, S / 128), T, 0, stream>>>(
      ctx_bf, 1024, wo, 1024, x2, 1024, S, 1024, 1024,
      nullptr, nullptr, nullptr, /*addSrc=*/x, nullptr, 0, one, 0, 0);
  copy_f32<<<(S * Dm) / 256, T, 0, stream>>>(x2, out, S * Dm);

  // 6) FFN RMSNorm + router + token compaction
  rmsnorm_bf16<<<S, T, 0, stream>>>(x2, ff_ln_w, xn2_bf, Dm);
  router_topk<<<S, T, 0, stream>>>(xn2_bf, router_w, dnsw);
  expert_assign<<<En, T, 0, stream>>>(dnsw, counts, idxb);

  // 7) MoE: gathered per-expert SwiGLU, gated scatter-accumulate into out
  for (int e = 0; e < En; ++e) {
    const int* Me   = counts + e;
    const int* idxe = idxb + (size_t)e * S;
    gemm_wmma<1><<<dim3(FFn / 128, S / 128), T, 0, stream>>>(
        xn2_bf, Dm, w1 + (size_t)e * Dm * FFn, FFn, gbuf, FFn, S, FFn, Dm,
        Me, idxe, nullptr, nullptr, nullptr, 0, one, 0, 0);
    gemm_wmma<1><<<dim3(FFn / 128, S / 128), T, 0, stream>>>(
        xn2_bf, Dm, w3 + (size_t)e * Dm * FFn, FFn, ubuf, FFn, S, FFn, Dm,
        Me, idxe, nullptr, nullptr, nullptr, 0, one, 0, 0);
    silu_mul<<<dim3(FFn / 256, S), T, 0, stream>>>(gbuf, ubuf, h_bf, Me);
    gemm_wmma<1><<<dim3(Dm / 128, S / 128), T, 0, stream>>>(
        h_bf, FFn, w2 + (size_t)e * FFn * Dm, Dm, out, Dm, S, Dm, FFn,
        Me, nullptr, idxe, nullptr, dnsw + e, En, one, /*accumulate=*/1, 0);
  }
}
